// Transformer_6519760355504
// MI455X (gfx1250) — matrix operand order repaired and checked by reading, not compiled, not run on hardware
//
#include <hip/hip_runtime.h>
#include <hip/hip_bf16.h>
#include <stdint.h>

// ---------------------------------------------------------------------------
// Transformer forward for MI455X (gfx1250), wave32 + WMMA bf16.
// GEMM: 128x64 block tile, 8 waves x 32x32 C each, double-buffered LDS
// (one barrier per 32-deep K-step), all staging/fragment traffic b128.
// ---------------------------------------------------------------------------

static constexpr int Bn  = 8;
static constexpr int Sn  = 512;
static constexpr int Tn  = 512;
static constexpr int Dn  = 512;
static constexpr int Hn  = 8;
static constexpr int Ln  = 6;
static constexpr int Fn  = 2048;
static constexpr int Vn  = 32000;
static constexpr int DKn = 64;      // D / H
static constexpr int BSn = Bn * Sn; // 4096 rows

typedef __attribute__((ext_vector_type(16))) __bf16 v16bf;
typedef __attribute__((ext_vector_type(8)))  float  v8f;
typedef unsigned short ushort8 __attribute__((ext_vector_type(8)));
typedef ushort8 ushort8a __attribute__((may_alias));

union FragBF {
    ushort8a       h[2];
    unsigned short u[16];
    v16bf          v;
};

__device__ __forceinline__ unsigned short f32_to_bf16_rne(float f) {
    unsigned int u = __float_as_uint(f);
    u += 0x7FFFu + ((u >> 16) & 1u);   // round-to-nearest-even
    return (unsigned short)(u >> 16);
}

// GEMM flags
#define GEMM_RELU       1
#define GEMM_WRITE_F32  2
#define GEMM_WRITE_BF16 4

// ---------------------------------------------------------------------------
// C[z] = relu(alpha * A[z] x B[z] + bias)
// BNMAJOR: B source stored N-major ([N][K] rows, pitch ldb) -> b128 staging
// !BNMAJOR: B source stored K-major ([K][N] rows, pitch ldb) -> scatter staging
// Batch offsets: off = (z / Hsub) * Outer + (z % Hsub) * Inner  (per matrix)
// M % 128 == 0, N % 64 == 0, K % 32 == 0 for every launch in this file.
// ---------------------------------------------------------------------------
template<bool BNMAJOR>
__global__ __launch_bounds__(256)
void gemm_bf16_wmma(const unsigned short* __restrict__ A,
                    const unsigned short* __restrict__ Bm,
                    const float* __restrict__ bias,
                    float* __restrict__ C32,
                    unsigned short* __restrict__ Cbf,
                    int K, int lda, int ldb, int ldc,
                    long aOuter, long aInner,
                    long bOuter, long bInner,
                    long cOuter, long cInner,
                    int Hsub, float alpha, int flags)
{
    // double-buffered tiles, fragment-major, 80 B row pitch (16B-aligned rows)
    __shared__ unsigned short As[2][128][40]; // [buf][M][K]
    __shared__ unsigned short Bs[2][64][40];  // [buf][N][K]

    const int z = blockIdx.z;
    A  += (long)(z / Hsub) * aOuter + (long)(z % Hsub) * aInner;
    Bm += (long)(z / Hsub) * bOuter + (long)(z % Hsub) * bInner;
    const long coff = (long)(z / Hsub) * cOuter + (long)(z % Hsub) * cInner;

    const int tid  = threadIdx.x;
    const int lane = tid & 31;
    const int wave = tid >> 5;
    const int wr   = wave >> 1;  // 0..3 : 32-row strip
    const int wc   = wave & 1;   // 0..1 : 32-col strip
    const int blockM = blockIdx.y * 128;
    const int blockN = blockIdx.x * 64;

    v8f acc00 = {0.f,0.f,0.f,0.f,0.f,0.f,0.f,0.f};
    v8f acc01 = {0.f,0.f,0.f,0.f,0.f,0.f,0.f,0.f};
    v8f acc10 = {0.f,0.f,0.f,0.f,0.f,0.f,0.f,0.f};
    v8f acc11 = {0.f,0.f,0.f,0.f,0.f,0.f,0.f,0.f};

    // staging coordinates (all b128-aligned)
    const int ar = tid >> 2;            // 0..63
    const int ac = (tid & 3) * 8;       // 0,8,16,24
    const int bk = tid >> 3;            // 0..31   (KMAJOR path)
    const int bn = (tid & 7) * 8;       // 0,8,..,56

    // fragment coordinates
    const int m0 = wr * 32 + (lane & 15);
    const int kh = (lane >> 4) * 8;     // A: K sub-group per lane-half
    const int nc = wc * 32 + (lane & 15);
    const int kb = (lane >> 4) * 8;     // B: K sub-group per lane-half

    const int nsteps = K >> 5;

    // ---- prologue: stage tile 0 into buffer 0 ----
    {
        *(ushort8a*)&As[0][ar][ac] =
            *(const ushort8a*)&A[(long)(blockM + ar) * lda + ac];
        *(ushort8a*)&As[0][64 + ar][ac] =
            *(const ushort8a*)&A[(long)(blockM + 64 + ar) * lda + ac];
        if (BNMAJOR) {
            *(ushort8a*)&Bs[0][ar][ac] =
                *(const ushort8a*)&Bm[(long)(blockN + ar) * ldb + ac];
        } else {
            ushort8a vb = *(const ushort8a*)&Bm[(long)bk * ldb + (blockN + bn)];
            #pragma unroll
            for (int j = 0; j < 8; ++j) Bs[0][bn + j][bk] = vb[j];
        }
    }
    __syncthreads();

    for (int i = 0; i < nsteps; ++i) {
        const int cur = i & 1;
        const int nxt = cur ^ 1;
        const bool hasNext = (i + 1) < nsteps;
        const int k0n = (i + 1) << 5;

        // issue next tile's global loads first (latency hidden under WMMAs)
        ushort8a va0, va1, vb;
        if (hasNext) {
            va0 = *(const ushort8a*)&A[(long)(blockM + ar) * lda + (k0n + ac)];
            va1 = *(const ushort8a*)&A[(long)(blockM + 64 + ar) * lda + (k0n + ac)];
            if (BNMAJOR)
                vb = *(const ushort8a*)&Bm[(long)(blockN + ar) * ldb + (k0n + ac)];
            else
                vb = *(const ushort8a*)&Bm[(long)(k0n + bk) * ldb + (blockN + bn)];
        }

        // fragments per ISA 7.12.2 (contiguous 16 B LDS reads) + 4 WMMAs
        FragBF a0, a1, b0, b1;
        a0.h[0] = *(const ushort8a*)&As[cur][m0][kh];
        a0.h[1] = *(const ushort8a*)&As[cur][m0][16 + kh];
        a1.h[0] = *(const ushort8a*)&As[cur][m0 + 16][kh];
        a1.h[1] = *(const ushort8a*)&As[cur][m0 + 16][16 + kh];
        b0.h[0] = *(const ushort8a*)&Bs[cur][nc][kb];
        b0.h[1] = *(const ushort8a*)&Bs[cur][nc][16 + kb];
        b1.h[0] = *(const ushort8a*)&Bs[cur][nc + 16][kb];
        b1.h[1] = *(const ushort8a*)&Bs[cur][nc + 16][16 + kb];

        acc00 = __builtin_amdgcn_wmma_f32_16x16x32_bf16(false, a0.v, false, b0.v,
                                                        (short)0, acc00, false, false);
        acc01 = __builtin_amdgcn_wmma_f32_16x16x32_bf16(false, a0.v, false, b1.v,
                                                        (short)0, acc01, false, false);
        acc10 = __builtin_amdgcn_wmma_f32_16x16x32_bf16(false, a1.v, false, b0.v,
                                                        (short)0, acc10, false, false);
        acc11 = __builtin_amdgcn_wmma_f32_16x16x32_bf16(false, a1.v, false, b1.v,
                                                        (short)0, acc11, false, false);

        // drain next tile into the other buffer
        if (hasNext) {
            *(ushort8a*)&As[nxt][ar][ac]      = va0;
            *(ushort8a*)&As[nxt][64 + ar][ac] = va1;
            if (BNMAJOR) {
                *(ushort8a*)&Bs[nxt][ar][ac] = vb;
            } else {
                #pragma unroll
                for (int j = 0; j < 8; ++j) Bs[nxt][bn + j][bk] = vb[j];
            }
        }
        __syncthreads();   // single barrier per K-step (ping-pong)
    }

    // C/D layout: lane -> col = lane&15 (+16 for second tile),
    //             row = 8*(lane>>4) + r
    const int rb = blockM + wr * 32 + (lane >> 4) * 8;
    const int n0 = blockN + wc * 32 + (lane & 15);
    const int n1 = n0 + 16;
    float bias0 = 0.f, bias1 = 0.f;
    if (bias) { bias0 = bias[n0]; bias1 = bias[n1]; }

    #pragma unroll
    for (int r = 0; r < 8; ++r) {
        float v00 = acc00[r] * alpha + bias0;
        float v01 = acc01[r] * alpha + bias1;
        float v10 = acc10[r] * alpha + bias0;
        float v11 = acc11[r] * alpha + bias1;
        if (flags & GEMM_RELU) {
            v00 = fmaxf(v00, 0.f); v01 = fmaxf(v01, 0.f);
            v10 = fmaxf(v10, 0.f); v11 = fmaxf(v11, 0.f);
        }
        const long o0 = coff + (long)(rb + r) * ldc;
        const long o1 = coff + (long)(rb + 16 + r) * ldc;
        if (flags & GEMM_WRITE_F32) {
            C32[o0 + n0] = v00; C32[o0 + n1] = v01;
            C32[o1 + n0] = v10; C32[o1 + n1] = v11;
        }
        if (flags & GEMM_WRITE_BF16) {
            Cbf[o0 + n0] = f32_to_bf16_rne(v00); Cbf[o0 + n1] = f32_to_bf16_rne(v01);
            Cbf[o1 + n0] = f32_to_bf16_rne(v10); Cbf[o1 + n1] = f32_to_bf16_rne(v11);
        }
    }
}

// ---------------------------------------------------------------------------
// Masked softmax over rows of [B*H, Sq, Sk] (Sk == 512), probs out as bf16.
// causal==0: valid(k) = tok[b*Sk + k] != 0               (key padding mask)
// causal==1: valid(k) = (k <= q) && tok[b*Sq + q] != 0   (query-pad & nopeak)
// ---------------------------------------------------------------------------
__global__ __launch_bounds__(256)
void softmax_mask_kernel(const float* __restrict__ scores,
                         unsigned short* __restrict__ probs,
                         const int* __restrict__ tok,
                         int Sq, int Sk, int H, int causal)
{
    const int row = blockIdx.x;      // over B*H*Sq
    const int q   = row % Sq;
    const int z   = row / Sq;
    const int b   = z / H;
    const float* s = scores + (long)row * Sk;
    unsigned short* p = probs + (long)row * Sk;

    __shared__ float red[256];
    const int tid = threadIdx.x;

    const bool rowValid = causal ? (tok[b * Sq + q] != 0) : true;

    float vals[2];
    float mx = -3.0e38f;
    #pragma unroll
    for (int i = 0; i < 2; ++i) {      // Sk == 512 == 2*256
        int k = tid + i * 256;
        bool valid = causal ? (rowValid && (k <= q)) : (tok[b * Sk + k] != 0);
        float v = valid ? s[k] : -1.0e9f;
        vals[i] = v;
        mx = fmaxf(mx, v);
    }
    red[tid] = mx; __syncthreads();
    for (int off = 128; off > 0; off >>= 1) {
        if (tid < off) red[tid] = fmaxf(red[tid], red[tid + off]);
        __syncthreads();
    }
    mx = red[0]; __syncthreads();

    float sum = 0.f;
    #pragma unroll
    for (int i = 0; i < 2; ++i) { vals[i] = __expf(vals[i] - mx); sum += vals[i]; }
    red[tid] = sum; __syncthreads();
    for (int off = 128; off > 0; off >>= 1) {
        if (tid < off) red[tid] += red[tid + off];
        __syncthreads();
    }
    const float inv = 1.f / red[0];
    #pragma unroll
    for (int i = 0; i < 2; ++i)
        p[tid + i * 256] = f32_to_bf16_rne(vals[i] * inv);
}

// ---------------------------------------------------------------------------
// out = LayerNorm(resid + add) * g + beta   (D == 512, in-place on resid OK)
// ---------------------------------------------------------------------------
__global__ __launch_bounds__(256)
void add_ln_kernel(const float* __restrict__ resid,
                   const float* __restrict__ add,
                   const float* __restrict__ g,
                   const float* __restrict__ beta,
                   float* __restrict__ out32,
                   unsigned short* __restrict__ outbf)
{
    const long base = (long)blockIdx.x * Dn;
    const int tid = threadIdx.x;
    __shared__ float rs[256];
    __shared__ float rq[256];

    const float x0 = resid[base + tid]       + add[base + tid];
    const float x1 = resid[base + tid + 256] + add[base + tid + 256];
    rs[tid] = x0 + x1;
    rq[tid] = x0 * x0 + x1 * x1;
    __syncthreads();
    for (int off = 128; off > 0; off >>= 1) {
        if (tid < off) { rs[tid] += rs[tid + off]; rq[tid] += rq[tid + off]; }
        __syncthreads();
    }
    const float mu  = rs[0] * (1.f / Dn);
    const float var = rq[0] * (1.f / Dn) - mu * mu;
    const float inv = rsqrtf(var + 1e-5f);

    const float y0 = (x0 - mu) * inv * g[tid]       + beta[tid];
    const float y1 = (x1 - mu) * inv * g[tid + 256] + beta[tid + 256];
    out32[base + tid]        = y0;
    out32[base + tid + 256]  = y1;
    outbf[base + tid]        = f32_to_bf16_rne(y0);
    outbf[base + tid + 256]  = f32_to_bf16_rne(y1);
}

// ---------------------------------------------------------------------------
// x = emb[token] + pos_enc[s]   -> f32 + bf16
// ---------------------------------------------------------------------------
__global__ __launch_bounds__(256)
void embed_kernel(const int* __restrict__ tok,
                  const float* __restrict__ emb,
                  const float* __restrict__ pos,
                  float* __restrict__ out32,
                  unsigned short* __restrict__ outbf,
                  int Slen, long n)
{
    long i = (long)blockIdx.x * 256 + threadIdx.x;
    if (i >= n) return;
    const int  d  = (int)(i % Dn);
    const long rs = i / Dn;
    const int  s  = (int)(rs % Slen);
    const float v = emb[(long)tok[rs] * Dn + d] + pos[(long)s * Dn + d];
    out32[i] = v;
    outbf[i] = f32_to_bf16_rne(v);
}

// ---------------------------------------------------------------------------
// Transposing f32 -> bf16 convert: out[z][n*K + k] = bf16(in[z][k*N + n])
// Tiled 32x32 through LDS; grid = (N/32, K/32, nmat); block = 256 (32x8).
// ---------------------------------------------------------------------------
__global__ __launch_bounds__(256)
void cvt_bf16_t_kernel(const float* __restrict__ in,
                       unsigned short* __restrict__ out, int K, int N)
{
    __shared__ float tile[32][33];
    const long zoff = (long)blockIdx.z * K * N;
    const int n0 = blockIdx.x * 32;
    const int k0 = blockIdx.y * 32;
    const int tx = threadIdx.x & 31;
    const int ty = threadIdx.x >> 5;   // 0..7
    #pragma unroll
    for (int i = 0; i < 32; i += 8)
        tile[ty + i][tx] = in[zoff + (long)(k0 + ty + i) * N + (n0 + tx)];
    __syncthreads();
    #pragma unroll
    for (int i = 0; i < 32; i += 8)
        out[zoff + (long)(n0 + ty + i) * K + (k0 + tx)] =
            f32_to_bf16_rne(tile[tx][ty + i]);
}

// ---------------------------------------------------------------------------
// Host-side launch helpers
// ---------------------------------------------------------------------------
static void launch_gemm(hipStream_t st, bool bnmajor,
                        const unsigned short* A, const unsigned short* Bm,
                        const float* bias, float* C32, unsigned short* Cbf,
                        int M, int N, int K, int lda, int ldb, int ldc,
                        int batch, int Hsub,
                        long aOuter, long aInner, long bOuter, long bInner,
                        long cOuter, long cInner,
                        float alpha, int flags)
{
    dim3 grid(N / 64, M / 128, batch);
    if (bnmajor)
        gemm_bf16_wmma<true><<<grid, 256, 0, st>>>(A, Bm, bias, C32, Cbf,
            K, lda, ldb, ldc, aOuter, aInner, bOuter, bInner, cOuter, cInner,
            Hsub, alpha, flags);
    else
        gemm_bf16_wmma<false><<<grid, 256, 0, st>>>(A, Bm, bias, C32, Cbf,
            K, lda, ldb, ldc, aOuter, aInner, bOuter, bInner, cOuter, cInner,
            Hsub, alpha, flags);
}

static void launch_cvt_t(hipStream_t st, const float* in, unsigned short* out,
                         int K, int N, int nmat)
{
    dim3 grid(N / 32, K / 32, nmat);
    cvt_bf16_t_kernel<<<grid, 256, 0, st>>>(in, out, K, N);
}

// One multi-head attention block. Wt = 4 consecutive transposed DxD bf16 mats.
// Result (bias applied, pre-residual) -> tmp32 f32 [B*S, D].
static void run_mha(hipStream_t st,
                    const unsigned short* qsrc, const unsigned short* kvsrc,
                    const unsigned short* Wt, const float* bvec,
                    const int* tok, int causal,
                    unsigned short* qbf, unsigned short* kbf, unsigned short* vbf,
                    float* scores32, unsigned short* probsbf,
                    unsigned short* ctxbf, float* tmp32)
{
    const int Sq = Sn, Sk = Sn;           // S == T == 512 throughout
    const long DD = (long)Dn * Dn;
    // Q, K, V projections (bias fused), bf16 outputs
    launch_gemm(st, true, qsrc,  Wt + 0 * DD, bvec + 0 * Dn, nullptr, qbf,
                BSn, Dn, Dn, Dn, Dn, Dn, 1, 1, 0,0, 0,0, 0,0, 1.f, GEMM_WRITE_BF16);
    launch_gemm(st, true, kvsrc, Wt + 1 * DD, bvec + 1 * Dn, nullptr, kbf,
                BSn, Dn, Dn, Dn, Dn, Dn, 1, 1, 0,0, 0,0, 0,0, 1.f, GEMM_WRITE_BF16);
    launch_gemm(st, true, kvsrc, Wt + 2 * DD, bvec + 2 * Dn, nullptr, vbf,
                BSn, Dn, Dn, Dn, Dn, Dn, 1, 1, 0,0, 0,0, 0,0, 1.f, GEMM_WRITE_BF16);
    // scores[z] = (Q_z x K_z^T) / sqrt(DK); K matrix is naturally N-major here
    launch_gemm(st, true, qbf, kbf, nullptr, scores32, nullptr,
                Sq, Sk, DKn, Dn, Dn, Sk,
                Bn * Hn, Hn,
                (long)Sn * Dn, DKn,
                (long)Sn * Dn, DKn,
                (long)Hn * Sq * Sk, (long)Sq * Sk,
                0.125f, GEMM_WRITE_F32);
    softmax_mask_kernel<<<Bn * Hn * Sq, 256, 0, st>>>(scores32, probsbf, tok,
                                                      Sq, Sk, Hn, causal);
    // ctx[z] = probs_z x V_z ; V is K-major -> scatter staging path
    launch_gemm(st, false, probsbf, vbf, nullptr, nullptr, ctxbf,
                Sq, DKn, Sk, Sk, Dn, Dn,
                Bn * Hn, Hn,
                (long)Hn * Sq * Sk, (long)Sq * Sk,
                (long)Sn * Dn, DKn,
                (long)Sq * Dn, DKn,
                1.f, GEMM_WRITE_BF16);
    // output projection (bias fused) -> tmp32
    launch_gemm(st, true, ctxbf, Wt + 3 * DD, bvec + 3 * Dn, tmp32, nullptr,
                BSn, Dn, Dn, Dn, Dn, Dn, 1, 1, 0,0, 0,0, 0,0, 1.f, GEMM_WRITE_F32);
}

extern "C" void kernel_launch(void* const* d_in, const int* in_sizes, int n_in,
                              void* d_out, int out_size, void* d_ws, size_t ws_size,
                              hipStream_t stream)
{
    (void)in_sizes; (void)n_in; (void)out_size; (void)ws_size;

    const int*   src        = (const int*)  d_in[0];
    const int*   tgt        = (const int*)  d_in[1];
    const float* enc_emb    = (const float*)d_in[2];
    const float* dec_emb    = (const float*)d_in[3];
    const float* pos_enc    = (const float*)d_in[4];
    const float* enc_attn_w = (const float*)d_in[5];
    const float* enc_attn_b = (const float*)d_in[6];
    const float* enc_ff_w1  = (const float*)d_in[7];
    const float* enc_ff_b1  = (const float*)d_in[8];
    const float* enc_ff_w2  = (const float*)d_in[9];
    const float* enc_ff_b2  = (const float*)d_in[10];
    const float* enc_ln_g   = (const float*)d_in[11];
    const float* enc_ln_b   = (const float*)d_in[12];
    const float* dec_attn_w = (const float*)d_in[13];
    const float* dec_attn_b = (const float*)d_in[14];
    const float* dec_ff_w1  = (const float*)d_in[15];
    const float* dec_ff_b1  = (const float*)d_in[16];
    const float* dec_ff_w2  = (const float*)d_in[17];
    const float* dec_ff_b2  = (const float*)d_in[18];
    const float* dec_ln_g   = (const float*)d_in[19];
    const float* dec_ln_b   = (const float*)d_in[20];
    const float* fc_w       = (const float*)d_in[21];
    const float* fc_b       = (const float*)d_in[22];
    float* out = (float*)d_out;

    // ---- workspace bump allocator ----
    char* wp = (char*)d_ws;
    auto alloc = [&](size_t bytes) {
        void* r = (void*)wp;
        wp += (bytes + 255) & ~(size_t)255;
        return r;
    };
    const long nEncAttnW = (long)Ln * 4 * Dn * Dn;
    const long nDecAttnW = (long)Ln * 8 * Dn * Dn;
    const long nFfW      = (long)Ln * Dn * Fn;
    const long nFcW      = (long)Dn * Vn;

    unsigned short* w_enc_attn = (unsigned short*)alloc(nEncAttnW * 2); // [mat][N][K]
    unsigned short* w_dec_attn = (unsigned short*)alloc(nDecAttnW * 2);
    unsigned short* w_enc_ff1  = (unsigned short*)alloc(nFfW * 2);      // [l][F][D]
    unsigned short* w_enc_ff2  = (unsigned short*)alloc(nFfW * 2);      // [l][D][F]
    unsigned short* w_dec_ff1  = (unsigned short*)alloc(nFfW * 2);
    unsigned short* w_dec_ff2  = (unsigned short*)alloc(nFfW * 2);
    unsigned short* w_fc       = (unsigned short*)alloc(nFcW * 2);      // [V][D]

    float*          ex32     = (float*)alloc((size_t)BSn * Dn * 4);
    unsigned short* exbf     = (unsigned short*)alloc((size_t)BSn * Dn * 2);
    float*          y32      = (float*)alloc((size_t)BSn * Dn * 4);
    unsigned short* ybf      = (unsigned short*)alloc((size_t)BSn * Dn * 2);
    unsigned short* qbf      = (unsigned short*)alloc((size_t)BSn * Dn * 2);
    unsigned short* kbf      = (unsigned short*)alloc((size_t)BSn * Dn * 2);
    unsigned short* vbf      = (unsigned short*)alloc((size_t)BSn * Dn * 2);
    unsigned short* ctxbf    = (unsigned short*)alloc((size_t)BSn * Dn * 2);
    float*          scores32 = (float*)alloc((size_t)Bn * Hn * Sn * Sn * 4);
    unsigned short* probsbf  = (unsigned short*)alloc((size_t)Bn * Hn * Sn * Sn * 2);
    unsigned short* hbf      = (unsigned short*)alloc((size_t)BSn * Fn * 2);
    float*          tmp32    = (float*)alloc((size_t)BSn * Dn * 4);

    // ---- weights: fp32 [K][N] -> bf16 [N][K] (transpose once per call) ----
    launch_cvt_t(stream, enc_attn_w, w_enc_attn, Dn, Dn, Ln * 4);
    launch_cvt_t(stream, dec_attn_w, w_dec_attn, Dn, Dn, Ln * 8);
    launch_cvt_t(stream, enc_ff_w1,  w_enc_ff1,  Dn, Fn, Ln);
    launch_cvt_t(stream, enc_ff_w2,  w_enc_ff2,  Fn, Dn, Ln);
    launch_cvt_t(stream, dec_ff_w1,  w_dec_ff1,  Dn, Fn, Ln);
    launch_cvt_t(stream, dec_ff_w2,  w_dec_ff2,  Fn, Dn, Ln);
    launch_cvt_t(stream, fc_w,       w_fc,       Dn, Vn, 1);

    const long nTok = (long)BSn * Dn;
    embed_kernel<<<(unsigned)((nTok + 255) / 256), 256, 0, stream>>>(
        src, enc_emb, pos_enc, ex32, exbf, Sn, nTok);

    // ---------------- encoder ----------------
    for (int l = 0; l < Ln; ++l) {
        const unsigned short* Wl = w_enc_attn + (long)l * 4 * Dn * Dn;
        const float* Bl = enc_attn_b + (long)l * 4 * Dn;
        run_mha(stream, exbf, exbf, Wl, Bl, src, /*causal=*/0,
                qbf, kbf, vbf, scores32, probsbf, ctxbf, tmp32);
        add_ln_kernel<<<BSn, 256, 0, stream>>>(
            ex32, tmp32, enc_ln_g + (long)(l * 2 + 0) * Dn,
            enc_ln_b + (long)(l * 2 + 0) * Dn, ex32, exbf);
        // FFN
        launch_gemm(stream, true, exbf, w_enc_ff1 + (long)l * Dn * Fn,
                    enc_ff_b1 + (long)l * Fn, nullptr, hbf,
                    BSn, Fn, Dn, Dn, Dn, Fn, 1, 1, 0,0, 0,0, 0,0,
                    1.f, GEMM_RELU | GEMM_WRITE_BF16);
        launch_gemm(stream, true, hbf, w_enc_ff2 + (long)l * Fn * Dn,
                    enc_ff_b2 + (long)l * Dn, tmp32, nullptr,
                    BSn, Dn, Fn, Fn, Fn, Dn, 1, 1, 0,0, 0,0, 0,0,
                    1.f, GEMM_WRITE_F32);
        add_ln_kernel<<<BSn, 256, 0, stream>>>(
            ex32, tmp32, enc_ln_g + (long)(l * 2 + 1) * Dn,
            enc_ln_b + (long)(l * 2 + 1) * Dn, ex32, exbf);
    }
    // exbf now holds enc_out (bf16) for cross attention.

    // ---------------- decoder ----------------
    embed_kernel<<<(unsigned)((nTok + 255) / 256), 256, 0, stream>>>(
        tgt, dec_emb, pos_enc, y32, ybf, Tn, nTok);

    for (int l = 0; l < Ln; ++l) {
        const unsigned short* Wl = w_dec_attn + (long)l * 8 * Dn * Dn;
        const float* Bl = dec_attn_b + (long)l * 8 * Dn;
        // masked self-attention
        run_mha(stream, ybf, ybf, Wl, Bl, tgt, /*causal=*/1,
                qbf, kbf, vbf, scores32, probsbf, ctxbf, tmp32);
        add_ln_kernel<<<BSn, 256, 0, stream>>>(
            y32, tmp32, dec_ln_g + (long)(l * 3 + 0) * Dn,
            dec_ln_b + (long)(l * 3 + 0) * Dn, y32, ybf);
        // cross-attention (K/V from encoder output)
        run_mha(stream, ybf, exbf, Wl + (long)4 * Dn * Dn, Bl + 4 * Dn,
                src, /*causal=*/0,
                qbf, kbf, vbf, scores32, probsbf, ctxbf, tmp32);
        add_ln_kernel<<<BSn, 256, 0, stream>>>(
            y32, tmp32, dec_ln_g + (long)(l * 3 + 1) * Dn,
            dec_ln_b + (long)(l * 3 + 1) * Dn, y32, ybf);
        // FFN
        launch_gemm(stream, true, ybf, w_dec_ff1 + (long)l * Dn * Fn,
                    dec_ff_b1 + (long)l * Fn, nullptr, hbf,
                    BSn, Fn, Dn, Dn, Dn, Fn, 1, 1, 0,0, 0,0, 0,0,
                    1.f, GEMM_RELU | GEMM_WRITE_BF16);
        launch_gemm(stream, true, hbf, w_dec_ff2 + (long)l * Fn * Dn,
                    dec_ff_b2 + (long)l * Dn, tmp32, nullptr,
                    BSn, Dn, Fn, Fn, Fn, Dn, 1, 1, 0,0, 0,0, 0,0,
                    1.f, GEMM_WRITE_F32);
        add_ln_kernel<<<BSn, 256, 0, stream>>>(
            y32, tmp32, dec_ln_g + (long)(l * 3 + 2) * Dn,
            dec_ln_b + (long)(l * 3 + 2) * Dn, y32, ybf);
    }

    // ------- final projection: [4096,512] x [512,32000] (w_fc is [V][D]) -------
    launch_gemm(stream, true, ybf, w_fc, fc_b, out, nullptr,
                BSn, Vn, Dn, Dn, Dn, Vn, 1, 1, 0,0, 0,0, 0,0,
                1.f, GEMM_WRITE_F32);
}
